// QuantizedLinear_75239237091685
// MI455X (gfx1250) — matrix operand order repaired and checked by reading, not compiled, not run on hardware
//
#include <hip/hip_runtime.h>
#include <hip/hip_bf16.h>

// ---- WMMA vector types (probe-verified signatures) ----
typedef __attribute__((ext_vector_type(16))) __bf16 v16bf;
typedef __attribute__((ext_vector_type(8)))  __bf16 v8bf;
typedef __attribute__((ext_vector_type(4)))  __bf16 v4bf;
typedef __attribute__((ext_vector_type(8)))  float  v8f;

#define M_DIM   8192        // B*S = 4*2048
#define N_DIM   11008       // OUT
#define K_DIM   4096        // IN
#define KWORDS  (K_DIM/2)   // packed words per weight row (one byte per int32)
#define NGROUPS 32          // K_DIM / 128
#define NT      (K_DIM/BK)  // 64 K-tiles
#define BM 128
#define BN 128
#define BK 64
#define LDA 72              // bf16 elems; 144B rows keep 16B alignment + kill conflicts
#define LDB 72

// two 16-byte LDS chunks -> one 16-element bf16 fragment
__device__ __forceinline__ v16bf ld_frag(const __bf16* p, int second_off) {
    v8bf lo = *reinterpret_cast<const v8bf*>(p);
    v8bf hi = *reinterpret_cast<const v8bf*>(p + second_off);
    return __builtin_shufflevector(lo, hi, 0,1,2,3,4,5,6,7,8,9,10,11,12,13,14,15);
}

__global__ __launch_bounds__(256, 2)
void qlin_wmma_kernel(const float* __restrict__ x,
                      const int*   __restrict__ wq,
                      const float* __restrict__ wscale,
                      const float* __restrict__ wzp,
                      float*       __restrict__ out) {
    __shared__ __bf16 sA[2][BM * LDA];   // x tile (bf16), double-buffered
    __shared__ __bf16 sB[2][BN * LDB];   // dequantized weight tile (bf16)

    const int tid   = threadIdx.x;
    const int lane  = tid & 31;
    const int wave  = tid >> 5;       // 0..7
    const int waveM = wave >> 2;      // 0..1  (64-row slice)
    const int waveN = wave & 3;       // 0..3  (32-col slice)
    const int blockM = blockIdx.y * BM;
    const int blockN = blockIdx.x * BN;

    const int lhalf = lane >> 4;      // K-half select per ISA fragment layout
    const int l15   = lane & 15;

    // A loader: 16 rows x 16 float4-cols per pass, 8 passes
    const int rA = tid >> 4;          // 0..15
    const int cA = (tid & 15) * 4;    // 0..60
    // B loader: 32 rows x 8 int4-word-groups per pass, 4 passes
    const int rB = tid >> 3;          // 0..31
    const int wB = (tid & 7) * 4;     // packed-word offset 0..28

    const float* xrow = x  + (size_t)(blockM + rA) * K_DIM + cA;
    const int*   wrow = wq + (size_t)(blockN + rB) * KWORDS + wB;

    v8f acc[4][2];
#pragma unroll
    for (int mi = 0; mi < 4; ++mi)
#pragma unroll
        for (int ni = 0; ni < 2; ++ni)
            acc[mi][ni] = (v8f){0.f,0.f,0.f,0.f,0.f,0.f,0.f,0.f};

    float4 aReg[8];
    int4   bReg[4];
    float  sReg[4], zReg[4];

    // ---- stage-in (global -> VGPR) for K-tile kt ----
    auto load_global = [&](int kt) {
#pragma unroll
        for (int p = 0; p < 8; ++p)
            aReg[p] = *reinterpret_cast<const float4*>(
                xrow + (size_t)p * 16 * K_DIM + (size_t)kt * BK);
        const int g = kt >> 1;        // 128-wide quant group; BK=64 stays inside
#pragma unroll
        for (int p = 0; p < 4; ++p) {
            bReg[p] = *reinterpret_cast<const int4*>(
                wrow + (size_t)p * 32 * KWORDS + (size_t)kt * (BK / 2));
            const int n = blockN + rB + p * 32;
            sReg[p] = wscale[n * NGROUPS + g];
            zReg[p] = wzp[n * NGROUPS + g];
        }
    };

    // ---- convert/dequant (VGPR -> LDS buf) ----
    auto store_lds = [&](int buf) {
#pragma unroll
        for (int p = 0; p < 8; ++p) {
            v4bf b;
            b[0] = (__bf16)aReg[p].x; b[1] = (__bf16)aReg[p].y;
            b[2] = (__bf16)aReg[p].z; b[3] = (__bf16)aReg[p].w;
            *reinterpret_cast<v4bf*>(&sA[buf][(rA + p * 16) * LDA + cA]) = b;
        }
#pragma unroll
        for (int p = 0; p < 4; ++p) {
            const float s = sReg[p], z = zReg[p];
            const int words[4] = {bReg[p].x, bReg[p].y, bReg[p].z, bReg[p].w};
            v8bf t;
#pragma unroll
            for (int j = 0; j < 4; ++j) {
                const int wrd = words[j];
                t[2*j]   = (__bf16)((float)((wrd >> 4) & 0xF) * s + z); // high nibble
                t[2*j+1] = (__bf16)((float)( wrd       & 0xF) * s + z); // then low
            }
            *reinterpret_cast<v8bf*>(&sB[buf][(rB + p * 32) * LDB + wB * 2]) = t;
        }
    };

    // ---- 2 K-steps of 16x16x32 bf16 WMMA from LDS buf ----
    auto compute = [&](int buf) {
#pragma unroll
        for (int ks = 0; ks < BK; ks += 32) {
            v16bf afr[4];
#pragma unroll
            for (int mi = 0; mi < 4; ++mi) {
                // A 16x32: lane<16 holds K[0..7]+K[16..23]; lane>=16 -> +8
                const __bf16* pa =
                    &sA[buf][(waveM * 64 + mi * 16 + l15) * LDA + ks + lhalf * 8];
                afr[mi] = ld_frag(pa, 16);
            }
            v16bf bfr[2];
#pragma unroll
            for (int ni = 0; ni < 2; ++ni) {
                // B 32x16: column N=lane&15, lane<16 holds K[0..7]+K[16..23]; lane>=16 -> +8
                const __bf16* pb =
                    &sB[buf][(waveN * 32 + ni * 16 + l15) * LDB + ks + lhalf * 8];
                bfr[ni] = ld_frag(pb, 16);
            }
#pragma unroll
            for (int mi = 0; mi < 4; ++mi)
#pragma unroll
                for (int ni = 0; ni < 2; ++ni)
                    acc[mi][ni] = __builtin_amdgcn_wmma_f32_16x16x32_bf16(
                        false, afr[mi], false, bfr[ni],
                        (short)0, acc[mi][ni], false, false);
        }
    };

    // ---------- double-buffered pipeline: one barrier per K-tile ----------
    load_global(0);
    store_lds(0);
    __syncthreads();

    for (int kt = 0; kt < NT; ++kt) {
        const int cur = kt & 1;
        if (kt + 1 < NT) {
            load_global(kt + 1);                 // global loads issue early
            __builtin_prefetch(xrow + (size_t)(kt + 2) * BK, 0, 3);
            __builtin_prefetch(wrow + (size_t)(kt + 2) * (BK / 2), 0, 3);
        }
        compute(cur);                            // WMMA from buf, overlaps loads
        if (kt + 1 < NT)
            store_lds(cur ^ 1);                  // fill other buffer
        __syncthreads();
    }

    // ---------- epilogue: C/D layout (VGPR r -> row r(+8), col lane&15) ----------
#pragma unroll
    for (int mi = 0; mi < 4; ++mi) {
#pragma unroll
        for (int ni = 0; ni < 2; ++ni) {
            const int m0 = blockM + waveM * 64 + mi * 16 + lhalf * 8;
            const int n0 = blockN + waveN * 32 + ni * 16 + l15;
            float* o = out + (size_t)m0 * N_DIM + n0;
#pragma unroll
            for (int r = 0; r < 8; ++r)
                o[(size_t)r * N_DIM] = acc[mi][ni][r];
        }
    }
}

extern "C" void kernel_launch(void* const* d_in, const int* in_sizes, int n_in,
                              void* d_out, int out_size, void* d_ws, size_t ws_size,
                              hipStream_t stream) {
    const float* x      = (const float*)d_in[0];
    const int*   wq     = (const int*)d_in[1];
    const float* wscale = (const float*)d_in[2];
    const float* wzp    = (const float*)d_in[3];
    float*       out    = (float*)d_out;

    dim3 grid(N_DIM / BN, M_DIM / BM);   // (86, 64)
    dim3 block(256);
    qlin_wmma_kernel<<<grid, block, 0, stream>>>(x, wq, wscale, wzp, out);
}
